// RecursiveNN_41961830482185
// MI455X (gfx1250) — compile-verified, matrix operand order checked by reading, not run on hardware
//
#include <hip/hip_runtime.h>

#define TREES   256
#define LEAVES  1024
#define EDIM    128
#define CDIM    5
#define THREADS 256
#define NWAVES  8

typedef __bf16 bf16_t;
typedef __attribute__((ext_vector_type(16))) __bf16 v16bf;
typedef __attribute__((ext_vector_type(8)))  __bf16 v8bf;
typedef __attribute__((ext_vector_type(8)))  float  v8f;
typedef __attribute__((ext_vector_type(4)))  float  v4f;
typedef __attribute__((ext_vector_type(4)))  unsigned u32x4;
typedef __attribute__((ext_vector_type(8)))  int      i32x8;
typedef __attribute__((ext_vector_type(4)))  int      i32x4;

// ---- LDS layout (bytes), total 266816 <= 320KB/WGP ----
#define OFF_BUF0  0         // ping frontier: 512 nodes x 128 bf16 = 131072
#define OFF_BUF1  131072    // pong frontier: 256 nodes x 128 bf16 = 65536
#define OFF_W     196608    // W bf16 [128][256] = 65536
#define OFF_P     262144    // P bf16 [16][128]  = 4096 (rows 5..15 zero)
#define OFF_BW    266240    // bW f32 [128] = 512
#define OFF_BP    266752    // bP f32 [16]  = 64
#define LDS_BYTES 266816

// TDM path: ROCm 7.2 (clang-22) 5-arg builtin per bridge doc; fall back to
// plain vector copies on any other toolchain / host pass.
#if defined(__HIP_DEVICE_COMPILE__) && (__clang_major__ < 23) && \
    __has_builtin(__builtin_amdgcn_tensor_load_to_lds) &&        \
    __has_builtin(__builtin_amdgcn_s_wait_tensorcnt)
#define USE_TDM 1
#else
#define USE_TDM 0
#endif

#define WMMA_BF16(af, bfr, cc)                                                 \
  __builtin_amdgcn_wmma_f32_16x16x32_bf16(false, (af), false, (bfr), (short)0, \
                                          (cc), false, false)

// Native fptrunc: lets the backend use packed f32->bf16 cvt (RNE in HW)
// instead of a 3-VALU-op manual rounding sequence per element.
__device__ __forceinline__ bf16_t f2bf(float f) { return (bf16_t)f; }

// A/B fragment for v_wmma_f32_16x16x32_bf16, row-major source with >=32
// contiguous bf16 per row. ISA layout: lane<16 -> K {0..7,16..23},
// lane>=16 -> K {8..15,24..31} of its row. Two ds_load_b128 per fragment.
__device__ __forceinline__ v16bf frag_lds(const bf16_t* row, int k0, int hl) {
  v8bf lo = *(const v8bf*)(row + k0 + hl * 8);
  v8bf hi = *(const v8bf*)(row + k0 + 16 + hl * 8);
  return __builtin_shufflevector(lo, hi, 0,1,2,3,4,5,6,7,8,9,10,11,12,13,14,15);
}

// Fragment sourced from global f32 leaves (read-once => non-temporal loads).
__device__ __forceinline__ v16bf frag_gf32(const float* row, int k0, int hl) {
  v4f a0 = __builtin_nontemporal_load((const v4f*)(row + k0 + hl * 8));
  v4f a1 = __builtin_nontemporal_load((const v4f*)(row + k0 + hl * 8 + 4));
  v4f b0 = __builtin_nontemporal_load((const v4f*)(row + k0 + 16 + hl * 8));
  v4f b1 = __builtin_nontemporal_load((const v4f*)(row + k0 + 16 + hl * 8 + 4));
  v16bf r;
#pragma unroll
  for (int i = 0; i < 4; ++i) {
    r[i]      = f2bf(a0[i]);
    r[4 + i]  = f2bf(a1[i]);
    r[8 + i]  = f2bf(b0[i]);
    r[12 + i] = f2bf(b1[i]);
  }
  return r;
}

#if USE_TDM
// 1D block copy global->LDS through the Tensor Data Mover.
// D# group0: count=1, lds_addr, global_addr, type=2.  group1: data_size=4B,
// tensor_dim0 = tile_dim0 = n_dw (dwords), tensor_dim1 = tile_dim1 = 1.
__device__ __forceinline__ void tdm_copy_dw(unsigned lds_off, const void* gsrc,
                                            unsigned n_dw) {
  unsigned long long ga = (unsigned long long)gsrc;
  u32x4 g0;
  g0[0] = 1u;                                           // count=1
  g0[1] = lds_off;                                      // lds_addr bytes
  g0[2] = (unsigned)ga;                                 // global_addr[31:0]
  g0[3] = (unsigned)((ga >> 32) & 0x01FFFFFFull) | (2u << 30);  // [56:32]|type=2
  i32x8 g1;
  g1[0] = (int)(2u << 16);                              // data_size = 4B
  g1[1] = (int)((n_dw & 0xFFFFu) << 16);                // tensor_dim0[15:0]
  g1[2] = (int)((n_dw >> 16) | (1u << 16));             // dim0[31:16], dim1=1
  g1[3] = (int)((n_dw & 0xFFFFu) << 16);                // tile_dim0 (fits 16b)
  g1[4] = 1;                                            // tile_dim1=1
  g1[5] = (int)n_dw;                                    // dim0_stride[31:0]
  g1[6] = 0;
  g1[7] = 0;
  i32x4 z = {0, 0, 0, 0};
  __builtin_amdgcn_tensor_load_to_lds(g0, g1, z, z, 0);
}
#endif

// Prologue: W [128][256] f32 -> bf16; P [5][128] f32 -> bf16 padded to [16][128].
__global__ void cvt_wp_kernel(const float* __restrict__ W, const float* __restrict__ P,
                              bf16_t* __restrict__ Wb, bf16_t* __restrict__ Pb) {
  int i = blockIdx.x * blockDim.x + threadIdx.x;
  if (i < EDIM * 2 * EDIM) Wb[i] = f2bf(W[i]);
  if (i < 16 * EDIM) {
    int r = i >> 7, c = i & 127;
    Pb[i] = (r < CDIM) ? f2bf(P[r * EDIM + c]) : f2bf(0.f);
  }
}

__global__ __launch_bounds__(THREADS)
void forest_kernel(const float* __restrict__ leaves,
                   const float* __restrict__ bW,
                   const float* __restrict__ bP,
                   const bf16_t* __restrict__ Wg,
                   const bf16_t* __restrict__ Pg,
                   float* __restrict__ out)
{
  extern __shared__ char smem[];
  bf16_t* buf0 = (bf16_t*)(smem + OFF_BUF0);
  bf16_t* buf1 = (bf16_t*)(smem + OFF_BUF1);
  bf16_t* Wl   = (bf16_t*)(smem + OFF_W);
  bf16_t* Pl   = (bf16_t*)(smem + OFF_P);
  float*  bWl  = (float*)(smem + OFF_BW);
  float*  bPl  = (float*)(smem + OFF_BP);

  const int tid  = threadIdx.x;
  const int b    = blockIdx.x;        // tree index
  const int lane = tid & 31;
  const int wave = tid >> 5;          // 0..7 == this wave's N-tile (fixed)
  const int hl   = lane >> 4;         // lane half (K-phase select)
  const int lidx = lane & 15;         // row (A) / col (B,D) index

  // ---- stage W/P into LDS: TDM if available, else 16B vector copies ----
#if USE_TDM
  if (wave == 0) {
    tdm_copy_dw((unsigned)(unsigned long long)(void*)Wl, Wg, 16384u); // 64KB
    tdm_copy_dw((unsigned)(unsigned long long)(void*)Pl, Pg, 1024u);  // 4KB
    __builtin_amdgcn_s_wait_tensorcnt((short)0);
  }
#else
  {
    const uint4* ws = (const uint4*)Wg;  uint4* wd = (uint4*)Wl;
#pragma unroll 1
    for (int i = tid; i < 4096; i += THREADS) wd[i] = ws[i];
    const uint4* ps = (const uint4*)Pg;  uint4* pd = (uint4*)Pl;
    for (int i = tid; i < 256; i += THREADS) pd[i] = ps[i];
  }
#endif
  if (tid < EDIM) bWl[tid] = bW[tid];
  if (tid < 16)   bPl[tid] = (tid < CDIM) ? bP[tid] : 0.f;
  __syncthreads();

  // ---- hoist loop-invariant B operands into VGPRs (held for whole kernel) ----
  // h-GEMM B: this wave's 16 columns of W, all 8 K-chunks  (64 VGPRs)
  // proj  B: P^T fragments, 4 K-chunks                     (32 VGPRs)
  const bf16_t* wrow = Wl + (wave * 16 + lidx) * 256;
  v16bf wb[8];
#pragma unroll
  for (int kk = 0; kk < 8; ++kk) wb[kk] = frag_lds(wrow, kk * 32, hl);
  const bf16_t* prow = Pl + lidx * EDIM;
  v16bf pb[4];
#pragma unroll
  for (int kk = 0; kk < 4; ++kk) pb[kk] = frag_lds(prow, kk * 32, hl);
  const int   hcol  = wave * 16 + lidx;      // this wave's output column
  const float hbias = bWl[hcol];
  const float pbias = (lidx < CDIM) ? bPl[lidx] : 0.f;

  const float* leaf_base = leaves + (size_t)b * (LEAVES * EDIM);

  // ================= level 0: A from global leaves (f32 -> bf16) ============
#pragma unroll 1
  for (int it = 0; it < 32; ++it) {          // 512 nodes / 16 rows
    const float* arow = leaf_base + (size_t)(it * 16 + lidx) * 256;
    __builtin_prefetch(arow + 16 * 256, 0, 0);  // next M-tile for this wave
    v8f acc = {};
#pragma unroll
    for (int kk = 0; kk < 8; ++kk)
      acc = WMMA_BF16(frag_gf32(arow, kk * 32, hl), wb[kk], acc);
#pragma unroll
    for (int r = 0; r < 8; ++r) {
      float v = acc[r] + hbias;
      v = v > 0.f ? v : 0.f;
      buf0[(it * 16 + hl * 8 + r) * EDIM + hcol] = f2bf(v);
    }
  }
  __syncthreads();

  // ================= levels 0..9 projections + levels 1..9 h ================
  size_t lvl_off = 0;
  const bf16_t* in_l  = buf0;
  bf16_t*       out_b = buf1;
  int n_out = 512;

#pragma unroll 1
  for (int lvl = 0; lvl < 10; ++lvl) {
    const int mtiles = (n_out + 15) >> 4;

    // ---- proj = h @ P^T + bP on frontier in in_l (N padded 5->16, K=128) ----
#pragma unroll 1
    for (int it = wave; it < mtiles; it += NWAVES) {
      const bf16_t* arow = in_l + (it * 16 + lidx) * EDIM;
      v8f acc = {};
#pragma unroll
      for (int kk = 0; kk < 4; ++kk)
        acc = WMMA_BF16(frag_lds(arow, kk * 32, hl), pb[kk], acc);
      if (lidx < CDIM) {
#pragma unroll
        for (int r = 0; r < 8; ++r) {
          const int row = it * 16 + hl * 8 + r;
          if (row < n_out)
            out[(lvl_off + (size_t)b * n_out + row) * CDIM + lidx] = acc[r] + pbias;
        }
      }
    }
    lvl_off += (size_t)TREES * n_out;
    if (lvl == 9) break;

    // ---- next frontier: h = relu(pairs @ W^T + bW), pairs row contiguous ----
    const int n_next = n_out >> 1;
    const int mt2 = (n_next + 15) >> 4;
#pragma unroll 1
    for (int it = 0; it < mt2; ++it) {
      const bf16_t* arow = in_l + (it * 16 + lidx) * 256;
      v8f acc = {};
#pragma unroll
      for (int kk = 0; kk < 8; ++kk)
        acc = WMMA_BF16(frag_lds(arow, kk * 32, hl), wb[kk], acc);
#pragma unroll
      for (int r = 0; r < 8; ++r) {
        float v = acc[r] + hbias;
        v = v > 0.f ? v : 0.f;
        out_b[(it * 16 + hl * 8 + r) * EDIM + hcol] = f2bf(v);
      }
    }
    __syncthreads();   // frontier writes visible before proj / next level reads

    in_l  = out_b;
    out_b = (out_b == buf0) ? buf1 : buf0;   // ping-pong (hazard-free)
    n_out = n_next;
  }
}

extern "C" void kernel_launch(void* const* d_in, const int* in_sizes, int n_in,
                              void* d_out, int out_size, void* d_ws, size_t ws_size,
                              hipStream_t stream) {
  const float* leaves = (const float*)d_in[0];
  const float* W      = (const float*)d_in[1];
  const float* bW     = (const float*)d_in[2];
  const float* P      = (const float*)d_in[3];
  const float* bP     = (const float*)d_in[4];
  float* out = (float*)d_out;

  bf16_t* Wb = (bf16_t*)d_ws;                       // 65536 B
  bf16_t* Pb = (bf16_t*)((char*)d_ws + 65536);      // 4096 B

  cvt_wp_kernel<<<(EDIM * 2 * EDIM + 255) / 256, 256, 0, stream>>>(W, P, Wb, Pb);

  (void)hipFuncSetAttribute((const void*)forest_kernel,
                            hipFuncAttributeMaxDynamicSharedMemorySize, LDS_BYTES);
  forest_kernel<<<TREES, THREADS, LDS_BYTES, stream>>>(leaves, bW, bP, Wb, Pb, out);
}
